// Attention_71399536329277
// MI455X (gfx1250) — compile-verified, matrix operand order checked by reading, not run on hardware
//
#include <hip/hip_runtime.h>
#include <hip/hip_bf16.h>
#include <math.h>

// Problem dims (from reference)
constexpr int Hd = 1024;            // hidden dim
constexpr int Sd = 1024;            // sequence length
constexpr int Bd = 32;              // batch
constexpr int Kd = 2 * Hd;          // GEMM K = 2048
constexpr int WSTRIDE = 3 * Hd;     // W_attn row stride = 3072
constexpr int Mtot = Sd * Bd;       // GEMM M = 32768

// GEMM tiling
constexpr int BM = 64;              // block tile M (4 m-tiles)
constexpr int BN = 128;             // block tile N (8 n-tiles)
constexpr int KSTEP = 64;           // K slab per LDS stage (2 wmma k-chunks)
constexpr int NK = Kd / KSTEP;      // 32 K iterations
constexpr int LSTRIDE = KSTEP + 8;  // padded LDS row stride (bf16 elems) -> conflict-free ds_load_b128
constexpr int A_ELEMS = BM * LSTRIDE;           // 4608
constexpr int B_ELEMS = BN * LSTRIDE;           // 9216
constexpr int BUF_ELEMS = A_ELEMS + B_ELEMS;    // 13824 bf16 per buffer

typedef __attribute__((ext_vector_type(16))) __bf16 v16bf;
typedef __attribute__((ext_vector_type(8)))  __bf16 v8bf;
typedef __attribute__((ext_vector_type(2)))  __bf16 v2bf;
typedef __attribute__((ext_vector_type(8)))  float  v8f;

__device__ __forceinline__ unsigned pk_bf16(float lo, float hi)
{
    v2bf v;
    v[0] = (__bf16)lo;
    v[1] = (__bf16)hi;
    return __builtin_bit_cast(unsigned, v);
}

__device__ __forceinline__ void load_chunk16(const float* __restrict__ src, float4 r[4])
{
    const float4* p = reinterpret_cast<const float4*>(src);
    r[0] = p[0]; r[1] = p[1]; r[2] = p[2]; r[3] = p[3];
}

// Convert 16 fp32 -> 16 bf16 and store 32B to LDS (16B aligned).
__device__ __forceinline__ void store_chunk_bf16(__bf16* dst, const float4 r[4])
{
    uint4* d = reinterpret_cast<uint4*>(dst);
    d[0] = make_uint4(pk_bf16(r[0].x, r[0].y), pk_bf16(r[0].z, r[0].w),
                      pk_bf16(r[1].x, r[1].y), pk_bf16(r[1].z, r[1].w));
    d[1] = make_uint4(pk_bf16(r[2].x, r[2].y), pk_bf16(r[2].z, r[2].w),
                      pk_bf16(r[3].x, r[3].y), pk_bf16(r[3].z, r[3].w));
}

// Read a 16-element bf16 WMMA operand fragment from LDS.
// CDNA5 16-bit operand layout per lane: lanes 0-15 take K=[0..7],[16..23];
// lanes 16-31 take K=[8..15],[24..31]  (kh = lane>>4).
__device__ __forceinline__ v16bf read_frag(const __bf16* base, int kh)
{
    v8bf lo = *reinterpret_cast<const v8bf*>(base + kh * 8);
    v8bf hi = *reinterpret_cast<const v8bf*>(base + 16 + kh * 8);
    return __builtin_shufflevector(lo, hi, 0, 1, 2, 3, 4, 5, 6, 7,
                                   8, 9, 10, 11, 12, 13, 14, 15);
}

// Kernel 1: h_proj[b,h] = hidden[b,:] . W_attn[h, 0:H] + b_attn[h]; also zero logits.
__global__ __launch_bounds__(256)
void hproj_init_kernel(const float* __restrict__ hidden,
                       const float* __restrict__ W,
                       const float* __restrict__ b_attn,
                       float* __restrict__ h_proj,
                       float* __restrict__ logits)
{
    int idx = blockIdx.x * blockDim.x + threadIdx.x;   // 0 .. 32767
    logits[idx] = 0.0f;
    int b = idx >> 10;
    int h = idx & (Hd - 1);
    const float* wrow = W + (size_t)h * WSTRIDE;       // W_h row h
    const float* hrow = hidden + (size_t)b * Hd;
    float s = 0.0f;
    #pragma unroll 4
    for (int k = 0; k < Hd; k += 4) {
        float4 wv = *reinterpret_cast<const float4*>(wrow + k);
        float4 hv = *reinterpret_cast<const float4*>(hrow + k);
        s = fmaf(wv.x, hv.x, s);
        s = fmaf(wv.y, hv.y, s);
        s = fmaf(wv.z, hv.z, s);
        s = fmaf(wv.w, hv.w, s);
    }
    h_proj[idx] = s + b_attn[h];
}

// Kernel 2 (main): fused e_proj GEMM (bf16 WMMA, fp32 accum) + tanh + v_w dot.
// Grid: (Hd/BN = 8, Mtot/BM = 512)  -- n-chunk FAST so the 8 WGs sharing one
// A-band run adjacently: enc streams from HBM once, reuse served by L2.
// Block: 256 threads = 8 waves. Wave w: m-tile = w>>1 (16 rows),
// n-half = w&1 (64 cols = 4 n-tiles).
__global__ __launch_bounds__(256)
void attn_energy_gemm(const float* __restrict__ enc,
                      const float* __restrict__ W,
                      const float* __restrict__ v_w,
                      const float* __restrict__ h_proj,
                      float* __restrict__ logits)
{
    __shared__ __bf16 smem[2 * BUF_ELEMS];   // 54 KB double-buffered A+B tiles

    const int t    = threadIdx.x;
    const int lane = t & 31;
    const int w    = t >> 5;
    const int lr   = lane & 15;
    const int kh   = lane >> 4;
    const int mt   = w >> 1;
    const int nh   = w & 1;

    const int mBlock = blockIdx.y * BM;
    const int nBlock = blockIdx.x * BN;
    const int mWave  = mBlock + mt * 16;
    const int nWave  = nBlock + nh * 64;

    // ---- staging assignment: 16-float chunks (4x float4, 64B coalesced) ----
    // A: 64 rows x 4 chunks = 256 -> one per thread.
    // B: 128 rows x 4 chunks = 512 -> two per thread (rows r and r+64).
    const int sRow = t >> 2;          // 0..63
    const int sQ   = t & 3;           // 16-float quarter within the 64-wide slab
    const float* gA  = enc + (size_t)(mBlock + sRow) * Kd + sQ * 16;
    const float* gB0 = W + (size_t)(nBlock + sRow) * WSTRIDE + Hd + sQ * 16;
    const float* gB1 = W + (size_t)(nBlock + sRow + 64) * WSTRIDE + Hd + sQ * 16;
    const int aDstOff  = sRow * LSTRIDE + sQ * 16;
    const int bDstOff0 = A_ELEMS + sRow * LSTRIDE + sQ * 16;
    const int bDstOff1 = A_ELEMS + (sRow + 64) * LSTRIDE + sQ * 16;

    // ---- per-wave fragment bases within a buffer ----
    const int aFragRow = (mt * 16 + lr) * LSTRIDE;
    const int bFragRow0 = A_ELEMS + (nh * 64 + 0 * 16 + lr) * LSTRIDE;
    const int bFragRow1 = A_ELEMS + (nh * 64 + 1 * 16 + lr) * LSTRIDE;
    const int bFragRow2 = A_ELEMS + (nh * 64 + 2 * 16 + lr) * LSTRIDE;
    const int bFragRow3 = A_ELEMS + (nh * 64 + 3 * 16 + lr) * LSTRIDE;

    v8f acc[4] = {};
    float4 ra[4], rb0[4], rb1[4];

    // prologue: stage K-slab 0
    load_chunk16(gA, ra);
    load_chunk16(gB0, rb0);
    load_chunk16(gB1, rb1);
    store_chunk_bf16(&smem[aDstOff], ra);
    store_chunk_bf16(&smem[bDstOff0], rb0);
    store_chunk_bf16(&smem[bDstOff1], rb1);
    __syncthreads();

    #pragma unroll 2
    for (int ks = 0; ks < NK; ++ks) {
        const int cur = ks & 1;
        const bool hasNext = (ks + 1 < NK);

        // issue next slab's global loads early; latency hidden by WMMAs below
        if (hasNext) {
            const int koff = (ks + 1) * KSTEP;
            load_chunk16(gA + koff, ra);
            load_chunk16(gB0 + koff, rb0);
            load_chunk16(gB1 + koff, rb1);
        }

        // compute current slab: 2 k-chunks x 4 n-tiles = 8 WMMAs
        const __bf16* buf = &smem[cur * BUF_ELEMS];
        #pragma unroll
        for (int kc = 0; kc < 2; ++kc) {
            v16bf af = read_frag(buf + aFragRow + kc * 32, kh);
            v16bf b0 = read_frag(buf + bFragRow0 + kc * 32, kh);
            v16bf b1 = read_frag(buf + bFragRow1 + kc * 32, kh);
            v16bf b2 = read_frag(buf + bFragRow2 + kc * 32, kh);
            v16bf b3 = read_frag(buf + bFragRow3 + kc * 32, kh);
            acc[0] = __builtin_amdgcn_wmma_f32_16x16x32_bf16(false, af, false, b0, (short)0, acc[0], false, false);
            acc[1] = __builtin_amdgcn_wmma_f32_16x16x32_bf16(false, af, false, b1, (short)0, acc[1], false, false);
            acc[2] = __builtin_amdgcn_wmma_f32_16x16x32_bf16(false, af, false, b2, (short)0, acc[2], false, false);
            acc[3] = __builtin_amdgcn_wmma_f32_16x16x32_bf16(false, af, false, b3, (short)0, acc[3], false, false);
        }

        // convert + store next slab into the other buffer
        if (hasNext) {
            __bf16* nbuf = &smem[(cur ^ 1) * BUF_ELEMS];
            store_chunk_bf16(nbuf + aDstOff, ra);
            store_chunk_bf16(nbuf + bDstOff0, rb0);
            store_chunk_bf16(nbuf + bDstOff1, rb1);
        }
        __syncthreads();
    }

    // Epilogue: energy = tanh(acc + h_proj[b(m), n]); partial = sum_n v_w[n]*energy.
    // C/D layout: VGPR r holds M=r (lanes 0-15) / M=r+8 (lanes 16-31), N = lr.
    float rowsum[8] = {0.f, 0.f, 0.f, 0.f, 0.f, 0.f, 0.f, 0.f};

    #pragma unroll
    for (int j = 0; j < 4; ++j) {
        int n = nWave + j * 16 + lr;
        float vw = v_w[n];
        #pragma unroll
        for (int r = 0; r < 8; ++r) {
            int m = mWave + r + kh * 8;                 // this lane's row for VGPR r
            float hp = h_proj[(m & (Bd - 1)) * Hd + n]; // b = m % B
            float e  = tanhf(acc[j][r] + hp);
            rowsum[r] = fmaf(vw, e, rowsum[r]);
        }
    }

    // Reduce over the 16 lanes of each half (N direction), then atomically
    // accumulate into logits[b*S + s] (m = s*B + b).
    #pragma unroll
    for (int r = 0; r < 8; ++r) {
        float v = rowsum[r];
        v += __shfl_xor(v, 1, 32);
        v += __shfl_xor(v, 2, 32);
        v += __shfl_xor(v, 4, 32);
        v += __shfl_xor(v, 8, 32);
        if (lr == 0) {
            int m = mWave + r + kh * 8;
            atomicAdd(&logits[(m & (Bd - 1)) * Sd + (m >> 5)], v);
        }
    }
}

// Kernel 3: softmax over S for each batch row. One block per b.
__global__ __launch_bounds__(256)
void softmax_rows(const float* __restrict__ logits, float* __restrict__ out)
{
    __shared__ float sm[256];
    const int b = blockIdx.x;
    const int t = threadIdx.x;
    const float* row = logits + (size_t)b * Sd;

    float lmax = -3.0e38f;
    for (int s = t; s < Sd; s += 256) lmax = fmaxf(lmax, row[s]);
    sm[t] = lmax;
    __syncthreads();
    for (int o = 128; o > 0; o >>= 1) {
        if (t < o) sm[t] = fmaxf(sm[t], sm[t + o]);
        __syncthreads();
    }
    const float mx = sm[0];
    __syncthreads();

    float lsum = 0.0f;
    for (int s = t; s < Sd; s += 256) lsum += __expf(row[s] - mx);
    sm[t] = lsum;
    __syncthreads();
    for (int o = 128; o > 0; o >>= 1) {
        if (t < o) sm[t] += sm[t + o];
        __syncthreads();
    }
    const float inv = 1.0f / sm[0];
    __syncthreads();

    for (int s = t; s < Sd; s += 256)
        out[(size_t)b * Sd + s] = __expf(row[s] - mx) * inv;
}

extern "C" void kernel_launch(void* const* d_in, const int* in_sizes, int n_in,
                              void* d_out, int out_size, void* d_ws, size_t ws_size,
                              hipStream_t stream)
{
    const float* enc    = (const float*)d_in[0];  // (S, B, 2H)
    const float* hidden = (const float*)d_in[1];  // (B, H)
    // d_in[2] = cell: unused by the reference
    const float* W      = (const float*)d_in[3];  // (H, 3H)
    const float* b_attn = (const float*)d_in[4];  // (H,)
    const float* v_w    = (const float*)d_in[5];  // (H,)
    float* out = (float*)d_out;                   // (B, S)

    float* h_proj = (float*)d_ws;                 // B*H fp32   = 128 KB
    float* logits = h_proj + Bd * Hd;             // B*S fp32   = 128 KB

    // 1) h_proj + zero logits
    hproj_init_kernel<<<(Bd * Hd) / 256, 256, 0, stream>>>(hidden, W, b_attn, h_proj, logits);

    // 2) fused bf16-WMMA GEMM + tanh + v-dot -> atomic logits
    //    grid.x = n-chunks (fast) so A-band reuse hits L2; grid.y = m-blocks
    attn_energy_gemm<<<dim3(Hd / BN, Mtot / BM), 256, 0, stream>>>(enc, W, v_w, h_proj, logits);

    // 3) softmax over S per batch row
    softmax_rows<<<Bd, 256, 0, stream>>>(logits, out);
}
